// AdaptiveSoftmax_24352464569077
// MI455X (gfx1250) — matrix operand order repaired and checked by reading, not compiled, not run on hardware
//
#include <hip/hip_runtime.h>
#include <cstddef>

typedef __attribute__((ext_vector_type(16))) __bf16 v16bf;
typedef __attribute__((ext_vector_type(8)))  float  v8f;

#define TOK 4096
#define NEG_BIG (-1.0e30f)
#define CHUNK 256   // columns per block: 8 waves x 2 subtiles x 16

// ---------------- WMMA fragment loaders (bf16 from f32 source) ----------------
// A fragment: 16x32 (MxK). lane r=lane&15 -> row m0+r; half=lane>>4.
// VGPR i (2 bf16 each): i<4 -> K = half*8 + 2i ; i>=4 -> K = 16 + half*8 + 2(i-4).
__device__ inline v16bf load_A_frag_f32(const float* __restrict__ A, int lda,
                                        int m0, int k0, int lane) {
  const int r = lane & 15;
  const int half = lane >> 4;
  const float* row = A + (size_t)(m0 + r) * lda + k0;
  v16bf a;
#pragma unroll
  for (int i = 0; i < 8; ++i) {
    const int kb = (i < 4) ? (half * 8 + 2 * i) : (16 + half * 8 + 2 * (i - 4));
    a[2 * i]     = (__bf16)row[kb];
    a[2 * i + 1] = (__bf16)row[kb + 1];
  }
  return a;
}

// B fragment: 32x16 (KxN) taken from row-major W[N][K] (B[k][n] = W[n][k]).
// lane r -> column n0+r; VGPR i (2 bf16 each): i<4 -> K = half*8 + 2i ; i>=4 -> K = 16 + half*8 + 2(i-4).
// Row index is CLAMPED to N-1 (branch-free): out-of-range output columns are
// masked to -inf in the softmax tile, and plain-GEMM launches have N % CHUNK == 0,
// so the clamped values never influence a live result.
__device__ inline v16bf load_B_frag_f32(const float* __restrict__ W, int ldw,
                                        int n0, int k0, int N, int lane) {
  const int r = lane & 15;
  const int half = lane >> 4;
  int n = n0 + r;
  n = (n < N) ? n : (N - 1);
  const float* row = W + (size_t)n * ldw + k0;
  v16bf b;
#pragma unroll
  for (int i = 0; i < 8; ++i) {
    const int kb = (i < 4) ? (half * 8 + 2 * i) : (16 + half * 8 + 2 * (i - 4));
    b[2 * i]     = (__bf16)row[kb];
    b[2 * i + 1] = (__bf16)row[kb + 1];
  }
  return b;
}

// ---------------- GEMM: C[M,N] = A[M,K] @ W[N,K]^T (f32 out) ----------------
// grid: (M/16, N/CHUNK); block: 256 threads = 8 waves;
// wave w owns cols [w*32, w*32+32): two 16-wide subtiles, two accumulators.
// GUARD=false: N is a multiple of CHUNK -> branch-free pointer-stepped stores.
template <bool GUARD>
__global__ void __launch_bounds__(256)
wmma_gemm_kernel(const float* __restrict__ A, const float* __restrict__ W,
                 float* __restrict__ C, int M, int N, int K) {
  const int m0 = blockIdx.x * 16;
  const int wave = threadIdx.x >> 5;
  const int lane = threadIdx.x & 31;
  const int n0 = blockIdx.y * CHUNK + wave * 32;
  v8f c0 = {}, c1 = {};
  for (int k0 = 0; k0 < K; k0 += 32) {
    v16bf a  = load_A_frag_f32(A, K, m0, k0, lane);
    v16bf b0 = load_B_frag_f32(W, K, n0,      k0, N, lane);
    v16bf b1 = load_B_frag_f32(W, K, n0 + 16, k0, N, lane);
    c0 = __builtin_amdgcn_wmma_f32_16x16x32_bf16(false, a, false, b0, (short)0,
                                                 c0, false, false);
    c1 = __builtin_amdgcn_wmma_f32_16x16x32_bf16(false, a, false, b1, (short)0,
                                                 c1, false, false);
  }
  const int col = lane & 15;
  const int half = lane >> 4;
  float* p0 = C + (size_t)(m0 + half * 8) * N + n0 + col;
  float* p1 = p0 + 16;
#pragma unroll
  for (int j = 0; j < 8; ++j) {
    if (!GUARD || n0 + col < N)      p0[0] = c0[j];
    if (!GUARD || n0 + 16 + col < N) p1[0] = c1[j];
    p0 += N;
    p1 += N;
  }
}

// ------------- fused logits tile + streaming softmax partials -------------
// logits tile = A[16 rows] @ W^T over a CHUNK-col slice (+bias), then per-row
// chunk-local (max, sum exp(x-max)) partials; label logit captured when the
// row's label falls in this chunk.
// mode 0: head (label = t<2000 ? t : (t<10000 ? 2000 : 2001))
// mode 1: tail (label = clamp(t - lo, 0, N-1))
__global__ void __launch_bounds__(256)
wmma_ce_partial_kernel(const float* __restrict__ A, const float* __restrict__ W,
                       const float* __restrict__ bias,
                       const long long* __restrict__ target,
                       int M, int N, int K, int nchunks, int mode, int lo,
                       float* __restrict__ partials,
                       float* __restrict__ label_logit) {
  __shared__ float tile[16][CHUNK + 1];
  __shared__ float red[16][17];
  __shared__ float redmax[16];

  const int m0 = blockIdx.x * 16;
  const int nchunk = blockIdx.y;
  const int wave = threadIdx.x >> 5;
  const int lane = threadIdx.x & 31;
  const int base = nchunk * CHUNK;
  const int n0 = base + wave * 32;

  v8f c0 = {}, c1 = {};
  for (int k0 = 0; k0 < K; k0 += 32) {
    v16bf a  = load_A_frag_f32(A, K, m0, k0, lane);
    v16bf b0 = load_B_frag_f32(W, K, n0,      k0, N, lane);
    v16bf b1 = load_B_frag_f32(W, K, n0 + 16, k0, N, lane);
    c0 = __builtin_amdgcn_wmma_f32_16x16x32_bf16(false, a, false, b0, (short)0,
                                                 c0, false, false);
    c1 = __builtin_amdgcn_wmma_f32_16x16x32_bf16(false, a, false, b1, (short)0,
                                                 c1, false, false);
  }

  const int col = lane & 15;
  const int half = lane >> 4;
  {
    const int gn0 = n0 + col;
    const int gn1 = n0 + 16 + col;
    const float bv0 = (bias != nullptr && gn0 < N) ? bias[gn0] : 0.0f;
    const float bv1 = (bias != nullptr && gn1 < N) ? bias[gn1] : 0.0f;
#pragma unroll
    for (int j = 0; j < 8; ++j) {
      const int row = half * 8 + j;
      tile[row][wave * 32 + col]      = (gn0 < N) ? (c0[j] + bv0) : NEG_BIG;
      tile[row][wave * 32 + 16 + col] = (gn1 < N) ? (c1[j] + bv1) : NEG_BIG;
    }
  }
  __syncthreads();

  const int row = threadIdx.x >> 4;  // 0..15
  const int sub = threadIdx.x & 15;  // 0..15

  float mx = NEG_BIG;
  for (int i = sub; i < CHUNK; i += 16) mx = fmaxf(mx, tile[row][i]);
  red[row][sub] = mx;
  __syncthreads();
  if (sub == 0) {
    float m2 = red[row][0];
#pragma unroll
    for (int i = 1; i < 16; ++i) m2 = fmaxf(m2, red[row][i]);
    redmax[row] = m2;
  }
  __syncthreads();
  const float rowmax = redmax[row];
  float s = 0.0f;
  for (int i = sub; i < CHUNK; i += 16) s += __expf(tile[row][i] - rowmax);
  __syncthreads();
  red[row][sub] = s;
  __syncthreads();
  if (sub == 0) {
    float ssum = 0.0f;
#pragma unroll
    for (int i = 0; i < 16; ++i) ssum += red[row][i];
    const int gm = m0 + row;
    partials[((size_t)gm * nchunks + nchunk) * 2 + 0] = rowmax;
    partials[((size_t)gm * nchunks + nchunk) * 2 + 1] = ssum;

    const long long t = target[gm];
    int label;
    if (mode == 0) {
      label = (t < 2000) ? (int)t : ((t < 10000) ? 2000 : 2001);
    } else {
      int l = (int)t - lo;
      label = l < 0 ? 0 : (l > N - 1 ? N - 1 : l);
    }
    if (label >= base && label < base + CHUNK)
      label_logit[gm] = tile[row][label - base];
  }
}

// ------------- combine chunk partials into per-row NLL -------------
__global__ void __launch_bounds__(256)
ce_row_reduce_kernel(const float* __restrict__ partials,
                     const float* __restrict__ label_logit,
                     const long long* __restrict__ target,
                     float* __restrict__ nll,
                     int M, int nchunks, int mode, int lo, int hi) {
  const int r = blockIdx.x * blockDim.x + threadIdx.x;
  if (r >= M) return;
  const float* p = partials + (size_t)r * nchunks * 2;
  float mx = NEG_BIG;
  for (int c = 0; c < nchunks; ++c) mx = fmaxf(mx, p[2 * c]);
  float s = 0.0f;
  for (int c = 0; c < nchunks; ++c) s += p[2 * c + 1] * __expf(p[2 * c] - mx);
  const float lse = mx + __logf(s);
  float w = 1.0f;
  if (mode == 1) {
    const long long t = target[r];
    w = (t >= (long long)lo && t < (long long)hi) ? 1.0f : 0.0f;
  }
  nll[r] = w * (lse - label_logit[r]);
}

// ------------- final scalar: mean over all NLL contributions -------------
__global__ void __launch_bounds__(256)
final_sum_kernel(const float* __restrict__ nll, int n, float scale,
                 float* __restrict__ out) {
  __shared__ float red[256];
  float s = 0.0f;
  for (int i = threadIdx.x; i < n; i += 256) s += nll[i];
  red[threadIdx.x] = s;
  __syncthreads();
  for (int off = 128; off > 0; off >>= 1) {
    if ((int)threadIdx.x < off) red[threadIdx.x] += red[threadIdx.x + off];
    __syncthreads();
  }
  if (threadIdx.x == 0) out[0] = red[0] * scale;
}

extern "C" void kernel_launch(void* const* d_in, const int* in_sizes, int n_in,
                              void* d_out, int out_size, void* d_ws, size_t ws_size,
                              hipStream_t stream) {
  (void)in_sizes; (void)n_in; (void)out_size; (void)ws_size;
  const float*     w_in   = (const float*)d_in[0];
  const long long* target = (const long long*)d_in[1];
  const float*     head_w = (const float*)d_in[2];  // [2002,1024]
  const float*     head_b = (const float*)d_in[3];  // [2002]
  const float*     t0w1   = (const float*)d_in[4];  // [1024,1024]
  const float*     t0w2   = (const float*)d_in[5];  // [8000,1024]
  const float*     t1w1   = (const float*)d_in[6];  // [256,1024]
  const float*     t1w2   = (const float*)d_in[7];  // [40000,256]

  const int M = TOK, K = 1024;
  const int N0 = 8000, N1 = 40000, NH = 2002;
  const int CH0 = (N0 + CHUNK - 1) / CHUNK;  // 32
  const int CH1 = (N1 + CHUNK - 1) / CHUNK;  // 157
  const int CHH = (NH + CHUNK - 1) / CHUNK;  // 8

  float* ws = (float*)d_ws;
  size_t off = 0;
  float* h0    = ws + off; off += (size_t)M * 1024;     // 4096x1024
  float* h1    = ws + off; off += (size_t)M * 256;      // 4096x256
  float* part0 = ws + off; off += (size_t)M * CH0 * 2;
  float* part1 = ws + off; off += (size_t)M * CH1 * 2;
  float* parth = ws + off; off += (size_t)M * CHH * 2;
  float* ll0   = ws + off; off += M;
  float* ll1   = ws + off; off += M;
  float* llh   = ws + off; off += M;
  float* nll   = ws + off; off += 3 * (size_t)M;        // [nll0 | nll1 | nllh]
  // total ~27 MB of workspace

  dim3 blk(256);

  // projections: h0 = w_in @ t0w1^T ; h1 = w_in @ t1w1^T  (N % CHUNK == 0)
  wmma_gemm_kernel<false><<<dim3(M / 16, 1024 / CHUNK), blk, 0, stream>>>(w_in, t0w1, h0, M, 1024, K);
  wmma_gemm_kernel<false><<<dim3(M / 16, 256 / CHUNK),  blk, 0, stream>>>(w_in, t1w1, h1, M, 256, K);

  // fused logits + softmax partials (logits never hit memory)
  wmma_ce_partial_kernel<<<dim3(M / 16, CH0), blk, 0, stream>>>(
      h0, t0w2, nullptr, target, M, N0, 1024, CH0, 1, 2000, part0, ll0);
  wmma_ce_partial_kernel<<<dim3(M / 16, CH1), blk, 0, stream>>>(
      h1, t1w2, nullptr, target, M, N1, 256, CH1, 1, 10000, part1, ll1);
  wmma_ce_partial_kernel<<<dim3(M / 16, CHH), blk, 0, stream>>>(
      w_in, head_w, head_b, target, M, NH, 1024, CHH, 0, 0, parth, llh);

  // per-row NLL
  ce_row_reduce_kernel<<<dim3(M / 256), blk, 0, stream>>>(part0, ll0, target, nll,         M, CH0, 1, 2000, 10000);
  ce_row_reduce_kernel<<<dim3(M / 256), blk, 0, stream>>>(part1, ll1, target, nll + M,     M, CH1, 1, 10000, 50000);
  ce_row_reduce_kernel<<<dim3(M / 256), blk, 0, stream>>>(parth, llh, target, nll + 2 * M, M, CHH, 0, 0, 0);

  final_sum_kernel<<<1, 256, 0, stream>>>(nll, 3 * M, 1.0f / (float)M, (float*)d_out);
}
